// MyLSTM_76295799046446
// MI455X (gfx1250) — matrix operand order repaired and checked by reading, not compiled, not run on hardware
//
#include <hip/hip_runtime.h>
#include <math.h>

// Problem constants (match reference).
#define BB 32      // batch
#define TT 250     // seq len
#define EE 256     // embed dim
#define UU 1024    // units
#define N4 4096    // 4*U (gate-concat width)

typedef __bf16 bf16;
typedef __attribute__((ext_vector_type(16))) __bf16 v16bf;
typedef __attribute__((ext_vector_type(8)))  __bf16 v8bf;
typedef __attribute__((ext_vector_type(8)))  float  v8f;

__device__ __forceinline__ bf16 f2bf(float f) {
  unsigned u = __builtin_bit_cast(unsigned, f);
  u += 0x7fffu + ((u >> 16) & 1u);          // round-to-nearest-even
  unsigned short h = (unsigned short)(u >> 16);
  return __builtin_bit_cast(bf16, h);
}
__device__ __forceinline__ float bf2f(bf16 b) {
  unsigned u = ((unsigned)__builtin_bit_cast(unsigned short, b)) << 16;
  return __builtin_bit_cast(float, u);
}
__device__ __forceinline__ float sigmoidf_(float z) { return 1.f / (1.f + __expf(-z)); }

// Load a 16x32 bf16 A-tile operand for this lane from a row-major [M x K] matrix.
// A layout (matches packed B): lane r (r=lane&15) holds row r; lanes<16 hold
// K={k0..k0+15}, lanes>=16 hold K={k0+16..k0+31}.
__device__ __forceinline__ v16bf loadA(const bf16* row, int k0, int half) {
  v8bf lo = *(const v8bf*)(row + k0 + half * 16);
  v8bf hi = *(const v8bf*)(row + k0 + half * 16 + 8);
  return __builtin_shufflevector(lo, hi, 0,1,2,3,4,5,6,7,8,9,10,11,12,13,14,15);
}

// ---------------------------------------------------------------------------
// Pack fp32 [K x N] weights into bf16 WMMA B-operand tile layout:
// tile (kt,nt) of shape 32x16; lane l holds 16 contiguous bf16 (32 B) where
// element j maps to (k = kt*32 + (l>>4)*16 + j, n = nt*16 + (l&15)).
// ---------------------------------------------------------------------------
__global__ void pack_b_kernel(const float* __restrict__ src, bf16* __restrict__ dst,
                              int K, int N) {
  const int Ntiles = N >> 4;
  const size_t total = (size_t)K * N;
  for (size_t p = (size_t)blockIdx.x * blockDim.x + threadIdx.x; p < total;
       p += (size_t)gridDim.x * blockDim.x) {
    int j = (int)(p & 15);
    size_t q = p >> 4;
    int l = (int)(q & 31);
    size_t tile = q >> 5;
    int nt = (int)(tile % Ntiles);
    int kt = (int)(tile / Ntiles);
    int k = kt * 32 + (l >> 4) * 16 + j;
    int n = nt * 16 + (l & 15);
    dst[p] = f2bf(src[(size_t)k * N + n]);
  }
}

// Embedding gather -> bf16 X laid out [t*B + b][E] (t-major so each timestep's
// 32 rows are contiguous and 16-row-tile aligned).
__global__ void embed_kernel(const int* __restrict__ tokens,
                             const float* __restrict__ emb,
                             bf16* __restrict__ X) {
  const size_t total = (size_t)TT * BB * EE;
  for (size_t p = (size_t)blockIdx.x * blockDim.x + threadIdx.x; p < total;
       p += (size_t)gridDim.x * blockDim.x) {
    int e = (int)(p & (EE - 1));
    size_t m = p >> 8;            // EE == 256
    int b = (int)(m & (BB - 1));
    int t = (int)(m >> 5);        // BB == 32
    int tok = tokens[b * TT + t];
    X[p] = f2bf(emb[(size_t)tok * EE + e]);
  }
}

// Generic WMMA GEMM: C[M x N] = A[M x K](bf16 row-major) * Bp(packed bf16) + bias[N].
// One 16x16 output tile per wave; grid = (M/16 * N/16)/4 blocks of 128 threads.
__global__ void gemm_bias_kernel(const bf16* __restrict__ A,
                                 const bf16* __restrict__ Bp,
                                 const float* __restrict__ bias,
                                 float* __restrict__ C,
                                 int M, int N, int K) {
  const int lane = threadIdx.x & 31;
  const int wave = blockIdx.x * (blockDim.x >> 5) + (threadIdx.x >> 5);
  const int Ntiles = N >> 4;
  const int mt = wave / Ntiles;
  const int nt = wave - mt * Ntiles;
  if (mt >= (M >> 4)) return;               // wave-uniform; EXEC stays all-ones
  const int half = lane >> 4, r = lane & 15;
  const bf16* arow = A + (size_t)(mt * 16 + r) * K;
  v8f acc = {};
  const int KT = K >> 5;
  for (int kt = 0; kt < KT; ++kt) {
    v16bf a = loadA(arow, kt * 32, half);
    v16bf b = *(const v16bf*)(Bp + (((size_t)kt * Ntiles + nt) * 32 + lane) * 16);
    acc = __builtin_amdgcn_wmma_f32_16x16x32_bf16(false, a, false, b,
                                                  (short)0, acc, false, false);
  }
  const float bv = bias[nt * 16 + r];
  for (int j = 0; j < 8; ++j) {
    int m = mt * 16 + half * 8 + j;         // C layout: VGPR j -> M = j + 8*half
    C[(size_t)m * N + nt * 16 + r] = acc[j] + bv;
  }
}

// ---------------------------------------------------------------------------
// Fused LSTM step: z = xk_t + h_in @ rk0 ; gates ; c_out = f*c+i*g ;
// h_out = o*tanh(c_out).
//
// Grid: 64 blocks x 512 threads (16 waves).  Block wg owns U-columns
// [wg*16, wg*16+16).  Wave w = (gate[1:0], mhalf, khalf): one 16x16 output
// tile (gate g, batch rows mhalf*16..+15) over half of K (khalf*512..+511),
// i.e. a 16-WMMA serial chain per wave (4x shorter than a full-K/2-tile wave,
// 4 waves/SIMD resident on a WGP for latency hiding).  K-halves are combined
// in LDS by the fused gate/state epilogue.  K cannot be split across *blocks*
// without a global sync, so 64 blocks (=U/16, each owning all 4 gates of its
// columns) is the max grid for a single fused kernel.
// ---------------------------------------------------------------------------
__global__ void lstm_step_kernel(const bf16* __restrict__ h_in,   // [32 x U] bf16 row-major
                                 const float* __restrict__ c_in,  // [32 x U]
                                 const bf16* __restrict__ Rp,     // packed rk [U x 4U]
                                 const float* __restrict__ xk_t,  // [32 x 4U]
                                 bf16* __restrict__ h_out,
                                 float* __restrict__ c_out) {
  __shared__ float zsh[4][2][32][16];       // gate x khalf x batch x col = 16 KB
  const int lane  = threadIdx.x & 31;
  const int wave  = threadIdx.x >> 5;       // 0..15
  const int khalf = wave & 1;
  const int mhalf = (wave >> 1) & 1;
  const int gate  = wave >> 2;
  const int wg = blockIdx.x;
  const int half = lane >> 4, r = lane & 15;
  const int Ntiles = N4 >> 4;               // 256
  const int nt = gate * (UU >> 4) + wg;     // N-tile index in packed rk space

  const bf16* arow = h_in + (size_t)(mhalf * 16 + r) * UU;
  v8f acc = {};
  const int kt0 = khalf * 16;               // 16 k-tiles of 32 per wave
  for (int kt = kt0; kt < kt0 + 16; ++kt) {
    v16bf b = *(const v16bf*)(Rp + (((size_t)kt * Ntiles + nt) * 32 + lane) * 16);
    v16bf a = loadA(arow, kt * 32, half);
    acc = __builtin_amdgcn_wmma_f32_16x16x32_bf16(false, a, false, b,
                                                  (short)0, acc, false, false);
  }
  // Deposit raw f32 partial tile: element j -> (m = mhalf*16 + half*8 + j, col = r).
  for (int j = 0; j < 8; ++j)
    zsh[gate][khalf][mhalf * 16 + half * 8 + j][r] = acc[j];
  __syncthreads();

  // Fused epilogue: 512 threads, one (batch, unit) cell each.
  {
    const int e = threadIdx.x;
    const int bm = e >> 4, col = e & 15;
    const int u = wg * 16 + col;
    const size_t xrow = (size_t)bm * N4 + wg * 16 + col;
    float zi = xk_t[xrow + 0 * UU] + zsh[0][0][bm][col] + zsh[0][1][bm][col];
    float zf = xk_t[xrow + 1 * UU] + zsh[1][0][bm][col] + zsh[1][1][bm][col];
    float zg = xk_t[xrow + 2 * UU] + zsh[2][0][bm][col] + zsh[2][1][bm][col];
    float zo = xk_t[xrow + 3 * UU] + zsh[3][0][bm][col] + zsh[3][1][bm][col];
    float gi = sigmoidf_(zi);
    float gf = sigmoidf_(zf);
    float gg = tanhf(zg);
    float go = sigmoidf_(zo);
    float c = gf * c_in[(size_t)bm * UU + u] + gi * gg;
    c_out[(size_t)bm * UU + u] = c;
    h_out[(size_t)bm * UU + u] = f2bf(go * tanhf(c));
  }
}

__global__ void zero_state_kernel(bf16* hz, float* cz) {
  int i = blockIdx.x * blockDim.x + threadIdx.x;
  if (i < BB * UU) { hz[i] = f2bf(0.f); cz[i] = 0.f; }
}

// out[b] = sigmoid( h0[b,:].wout[0:U] + h1[b,:].wout[U:2U] + bout )
__global__ void out_kernel(const bf16* __restrict__ h0, const bf16* __restrict__ h1,
                           const float* __restrict__ wout, const float* __restrict__ bout,
                           float* __restrict__ out) {
  __shared__ float red[256];
  const int b = blockIdx.x, tid = threadIdx.x;
  float s = 0.f;
  for (int u = tid; u < UU; u += 256) {
    s += bf2f(h0[(size_t)b * UU + u]) * wout[u];
    s += bf2f(h1[(size_t)b * UU + u]) * wout[UU + u];
  }
  red[tid] = s;
  __syncthreads();
  for (int off = 128; off > 0; off >>= 1) {
    if (tid < off) red[tid] += red[tid + off];
    __syncthreads();
  }
  if (tid == 0) out[b] = sigmoidf_(red[0] + bout[0]);
}

// ---------------------------------------------------------------------------
extern "C" void kernel_launch(void* const* d_in, const int* in_sizes, int n_in,
                              void* d_out, int out_size, void* d_ws, size_t ws_size,
                              hipStream_t stream) {
  const int*   tokens = (const int*)  d_in[0];
  const float* emb    = (const float*)d_in[1];
  const float* k0     = (const float*)d_in[2];
  const float* rk0    = (const float*)d_in[3];
  const float* b0     = (const float*)d_in[4];
  const float* k1     = (const float*)d_in[5];
  // d_in[6] = rk1: multiplied by h==0 in the reference -> provably unused.
  const float* b1     = (const float*)d_in[7];
  const float* wout   = (const float*)d_in[8];
  const float* bout   = (const float*)d_in[9];
  float* out = (float*)d_out;

  // Workspace carve-up (all offsets 256B-aligned by construction).
  char* w = (char*)d_ws;
  size_t off = 0;
  auto take = [&](size_t bytes) { void* p = w + off; off += (bytes + 255) & ~(size_t)255; return p; };
  bf16*  X    = (bf16*) take((size_t)TT * BB * EE * sizeof(bf16));   // 4.10 MB
  bf16*  K0P  = (bf16*) take((size_t)EE * N4 * sizeof(bf16));        // 2.10 MB
  bf16*  K1P  = (bf16*) take((size_t)EE * N4 * sizeof(bf16));        // 2.10 MB
  bf16*  RK0P = (bf16*) take((size_t)UU * N4 * sizeof(bf16));        // 8.39 MB
  float* XK   = (float*)take((size_t)TT * BB * N4 * sizeof(float));  // 131 MB
  float* XK1  = (float*)take((size_t)BB * N4 * sizeof(float));
  bf16*  HZ   = (bf16*) take((size_t)BB * UU * sizeof(bf16));
  float* CZ   = (float*)take((size_t)BB * UU * sizeof(float));
  bf16*  HA   = (bf16*) take((size_t)BB * UU * sizeof(bf16));
  bf16*  HB   = (bf16*) take((size_t)BB * UU * sizeof(bf16));
  float* CA   = (float*)take((size_t)BB * UU * sizeof(float));
  float* CB   = (float*)take((size_t)BB * UU * sizeof(float));
  bf16*  H1   = (bf16*) take((size_t)BB * UU * sizeof(bf16));
  float* C1   = (float*)take((size_t)BB * UU * sizeof(float));
  (void)ws_size; (void)n_in; (void)in_sizes; (void)out_size;

  // 1) Pack weights into WMMA B-operand layout (bf16).
  pack_b_kernel<<<4096, 256, 0, stream>>>(rk0, RK0P, UU, N4);
  pack_b_kernel<<<1024, 256, 0, stream>>>(k0,  K0P,  EE, N4);
  pack_b_kernel<<<1024, 256, 0, stream>>>(k1,  K1P,  EE, N4);

  // 2) Embedding gather -> bf16 X [t*B+b][E].
  embed_kernel<<<2000, 256, 0, stream>>>(tokens, emb, X);

  // 3) Zero initial state.
  zero_state_kernel<<<(BB * UU + 255) / 256, 256, 0, stream>>>(HZ, CZ);

  // 4) Hoisted input projections: XK = X@k0 + b0 (all T), XK1 = x_last@k1 + b1.
  {
    int tiles = (TT * BB / 16) * (N4 / 16);          // 500*256
    gemm_bias_kernel<<<tiles / 4, 128, 0, stream>>>(X, K0P, b0, XK, TT * BB, N4, EE);
  }
  {
    int tiles = (BB / 16) * (N4 / 16);               // 2*256
    gemm_bias_kernel<<<tiles / 4, 128, 0, stream>>>(
        X + (size_t)(TT - 1) * BB * EE, K1P, b1, XK1, BB, N4, EE);
  }

  // 5) Recurrent scan: 250 fused step kernels (stream order carries the dep).
  const bf16* hin = HZ; const float* cin = CZ;
  for (int t = 0; t < TT; ++t) {
    bf16*  hout = (t & 1) ? HB : HA;
    float* cout = (t & 1) ? CB : CA;
    lstm_step_kernel<<<UU / 16, 512, 0, stream>>>(hin, cin, RK0P,
                                                  XK + (size_t)t * BB * N4,
                                                  hout, cout);
    hin = hout; cin = cout;
  }

  // 6) Cell 1: single step with zero state (h_in=0 makes the R-term vanish,
  //    so reusing RK0P is exact).
  lstm_step_kernel<<<UU / 16, 512, 0, stream>>>(HZ, CZ, RK0P, XK1, H1, C1);

  // 7) Output head.
  out_kernel<<<BB, 256, 0, stream>>>(hin, H1, wout, bout, out);
}
